// quantized_TimesAttention3D_70703751627126
// MI455X (gfx1250) — 16-bit and 8-bit matrix operand order repaired and checked by reading, compiled, not run on hardware
//
#include <hip/hip_runtime.h>

typedef __attribute__((ext_vector_type(16))) _Float16 v16h;
typedef __attribute__((ext_vector_type(8)))  _Float16 v8h;
typedef __attribute__((ext_vector_type(8)))  float    v8f;
typedef __attribute__((ext_vector_type(8)))  int      v8i;
typedef __attribute__((ext_vector_type(4)))  int      v4i;
typedef __attribute__((ext_vector_type(2)))  int      v2i;

#define Bq 8
#define Nq 1024
#define DIMq 1024
#define Cq 512
#define Hq 16
#define HDq 32
#define O3q 1536
#define SCALEq 0.17677669529663687f

__global__ void k_cvt_f16(const float* __restrict__ in, _Float16* __restrict__ out, int n) {
  for (int i = blockIdx.x * blockDim.x + threadIdx.x; i < n; i += gridDim.x * blockDim.x)
    out[i] = (_Float16)in[i];
}

__global__ void k_quant_w(const float* __restrict__ w, const float* __restrict__ alpha,
                          _Float16* __restrict__ out, int O, int K) {
  int total = O * K;
  for (int i = blockIdx.x * blockDim.x + threadIdx.x; i < total; i += gridDim.x * blockDim.x) {
    int o = i / K;
    float a = alpha[o];
    float q = rintf(fminf(fmaxf(w[i] / a, -8.f), 7.f));
    out[i] = (_Float16)(q * a);
  }
}

__global__ void k_gemm_f16(const _Float16* __restrict__ A, const _Float16* __restrict__ W,
                           const float* __restrict__ bias, float* __restrict__ Cout,
                           int M, int Nc, int K) {
  const int lane = threadIdx.x;
  const int lm   = lane & 15;
  const int half = lane >> 4;
  const int nb = (blockIdx.x * 4 + threadIdx.y) * 32;
  const int mb = blockIdx.y * 16;

  const _Float16* arow = A + (size_t)(mb + lm) * K;
  const _Float16* w0   = W + (size_t)(nb + lm) * K;
  const _Float16* w1   = W + (size_t)(nb + 16 + lm) * K;
  v8f acc0 = {}, acc1 = {};
  for (int kb = 0; kb < K; kb += 32) {
    __builtin_prefetch(arow + kb + 256, 0, 1);
    v16h af = *(const v16h*)(arow + kb + 16 * half);
    v16h bf0 = *(const v16h*)(w0 + kb + 16 * half);
    v16h bf1 = *(const v16h*)(w1 + kb + 16 * half);
    acc0 = __builtin_amdgcn_wmma_f32_16x16x32_f16(false, af, false, bf0, (short)0, acc0,
                                                  false, false);
    acc1 = __builtin_amdgcn_wmma_f32_16x16x32_f16(false, af, false, bf1, (short)0, acc1,
                                                  false, false);
  }
  const float bv0 = bias ? bias[nb + lm] : 0.f;
  const float bv1 = bias ? bias[nb + 16 + lm] : 0.f;
#pragma unroll
  for (int r = 0; r < 8; ++r) {
    int row = r + 8 * half;
    Cout[(size_t)(mb + row) * Nc + nb + lm]      = acc0[r] + bv0;
    Cout[(size_t)(mb + row) * Nc + nb + 16 + lm] = acc1[r] + bv1;
  }
}

__global__ void k_qkv_quant(const float* __restrict__ qkv,
                            const float* __restrict__ bias_q, const float* __restrict__ bias_k,
                            const float* __restrict__ qa, const float* __restrict__ qz,
                            const float* __restrict__ ka, const float* __restrict__ kz,
                            const float* __restrict__ va, const float* __restrict__ vz,
                            signed char* __restrict__ q8, signed char* __restrict__ k8,
                            _Float16* __restrict__ vt) {
  int idx = blockIdx.x * blockDim.x + threadIdx.x;
  if (idx >= Bq * Hq * Nq) return;
  int n = idx % Nq, h = (idx / Nq) % Hq, b = idx / (Nq * Hq);
  const float aq = qa[h], zq = rintf(qz[h]);
  const float ak = ka[h], zk = rintf(kz[h]);
  const float av = va[h], zv = rintf(vz[h]);
  const float bqv = bias_q[h], bkv = bias_k[h];
  size_t base  = ((size_t)(b * Nq + n)) * O3q + h * HDq;
  size_t ob    = (size_t)idx * HDq;
  size_t vbase = (size_t)(b * Hq + h) * HDq * Nq + n;
#pragma unroll 4
  for (int d = 0; d < HDq; ++d) {
    float q = qkv[base + d] + bqv;
    float k = qkv[base + Cq + d] + bkv;
    float v = qkv[base + 2 * Cq + d];
    float qi = rintf(fminf(fmaxf(q / aq + zq, -8.f), 7.f)) - zq;
    float ki = rintf(fminf(fmaxf(k / ak + zk, -8.f), 7.f)) - zk;
    float vq = (rintf(fminf(fmaxf(v / av + zv, -8.f), 7.f)) - zv) * av;
    q8[ob + d] = (signed char)qi;
    k8[ob + d] = (signed char)ki;
    vt[vbase + (size_t)d * Nq] = (_Float16)vq;
  }
}

__global__ void k_attn(const signed char* __restrict__ Q, const signed char* __restrict__ Km,
                       const _Float16* __restrict__ Vt,
                       const float* __restrict__ qa, const float* __restrict__ ka,
                       const float* __restrict__ aa, const float* __restrict__ az,
                       _Float16* __restrict__ AO) {
  __shared__ __align__(32) _Float16 p_tile[4][16][32];
  const int lane = threadIdx.x, lm = lane & 15, half = lane >> 4;
  const int wv = threadIdx.y;
  const int qt = (blockIdx.x & 15) * 4 + wv;
  const int bh = blockIdx.x >> 4;
  const int h  = bh & 15;
  const float lsc = qa[h] * ka[h] * SCALEq;
  const float aalpha = aa[h], azp = rintf(az[h]);
  const signed char* Qb  = Q  + ((size_t)bh * Nq + qt * 16) * HDq;
  const signed char* Kb  = Km + (size_t)bh * Nq * HDq;
  const _Float16*    Vb0 = Vt + ((size_t)bh * HDq + lm) * Nq;
  const _Float16*    Vb1 = Vt + ((size_t)bh * HDq + 16 + lm) * Nq;

  v8i qf = {};
  {
    v2i qlo = *(const v2i*)(Qb + lm * HDq + 16 * half);
    v2i qhi = *(const v2i*)(Qb + lm * HDq + 16 * half + 8);
    qf[0] = qlo[0]; qf[1] = qlo[1]; qf[2] = qhi[0]; qf[3] = qhi[1];
  }

  float mrow[8], lrow[8];
#pragma unroll
  for (int r = 0; r < 8; ++r) { mrow[r] = -1e30f; lrow[r] = 0.f; }

  for (int kt = 0; kt < 64; ++kt) {
    v4i kw = *(const v4i*)(Kb + (kt * 16 + lm) * HDq + 16 * half);
    v8i kf = {};
    kf[0] = kw[0]; kf[1] = kw[1]; kf[2] = kw[2]; kf[3] = kw[3];
    v8i z = {};
    v8i s = __builtin_amdgcn_wmma_i32_16x16x64_iu8(true, qf, true, kf, z, false, false);
#pragma unroll
    for (int r = 0; r < 8; ++r) {
      float sv = (float)s[r] * lsc;
      float tm = sv;
      for (int o = 1; o < 16; o <<= 1) tm = fmaxf(tm, __shfl_xor(tm, o, 16));
      float mn = fmaxf(mrow[r], tm);
      float p = __expf(sv - mn);
      float ts = p;
      for (int o = 1; o < 16; o <<= 1) ts += __shfl_xor(ts, o, 16);
      lrow[r] = lrow[r] * __expf(mrow[r] - mn) + ts;
      mrow[r] = mn;
    }
  }
#pragma unroll
  for (int r = 0; r < 8; ++r) lrow[r] = 1.0f / lrow[r];

  v8f o0 = {}, o1 = {};
  for (int kt2 = 0; kt2 < 32; ++kt2) {
    const int k0 = kt2 * 32;
#pragma unroll
    for (int sub = 0; sub < 2; ++sub) {
      v4i kw = *(const v4i*)(Kb + (k0 + sub * 16 + lm) * HDq + 16 * half);
      v8i kf = {};
      kf[0] = kw[0]; kf[1] = kw[1]; kf[2] = kw[2]; kf[3] = kw[3];
      v8i z = {};
      v8i s = __builtin_amdgcn_wmma_i32_16x16x64_iu8(true, qf, true, kf, z, false, false);
#pragma unroll
      for (int r = 0; r < 8; ++r) {
        int row = r + 8 * half;
        float p = __expf((float)s[r] * lsc - mrow[r]) * lrow[r];
        float pq = (rintf(fminf(fmaxf(p / aalpha + azp, -128.f), 127.f)) - azp) * aalpha;
        p_tile[wv][row][sub * 16 + lm] = (_Float16)pq;
      }
    }
    v16h pf = *(const v16h*)&p_tile[wv][lm][16 * half];
    v16h b0 = *(const v16h*)(Vb0 + k0 + 16 * half);
    v16h b1 = *(const v16h*)(Vb1 + k0 + 16 * half);
    o0 = __builtin_amdgcn_wmma_f32_16x16x32_f16(false, pf, false, b0, (short)0, o0, false, false);
    o1 = __builtin_amdgcn_wmma_f32_16x16x32_f16(false, pf, false, b1, (short)0, o1, false, false);
  }

  const int b = bh >> 4;
#pragma unroll
  for (int r = 0; r < 8; ++r) {
    int row = r + 8 * half;
    size_t off = ((size_t)b * Nq + qt * 16 + row) * Cq + h * HDq;
    AO[off + lm]      = (_Float16)o0[r];
    AO[off + 16 + lm] = (_Float16)o1[r];
  }
}

extern "C" void kernel_launch(void* const* d_in, const int* in_sizes, int n_in,
                              void* d_out, int out_size, void* d_ws, size_t ws_size,
                              hipStream_t stream) {
  const float* x          = (const float*)d_in[0];
  const float* qkv_w      = (const float*)d_in[1];
  const float* qkv_alpha  = (const float*)d_in[2];
  const float* bias_q     = (const float*)d_in[3];
  const float* bias_k     = (const float*)d_in[4];
  const float* q_alpha    = (const float*)d_in[5];
  const float* q_zp       = (const float*)d_in[6];
  const float* k_alpha    = (const float*)d_in[7];
  const float* k_zp       = (const float*)d_in[8];
  const float* v_alpha    = (const float*)d_in[9];
  const float* v_zp       = (const float*)d_in[10];
  const float* attn_alpha = (const float*)d_in[11];
  const float* attn_zp    = (const float*)d_in[12];
  const float* proj_w     = (const float*)d_in[13];
  const float* proj_b     = (const float*)d_in[14];
  const float* proj_alpha = (const float*)d_in[15];

  char* ws = (char*)d_ws;
  size_t off = 0;
  _Float16* x_h  = (_Float16*)(ws + off); off += (size_t)Bq * Nq * DIMq * 2;
  _Float16* wq_h = (_Float16*)(ws + off); off += (size_t)O3q * DIMq * 2;
  _Float16* wp_h = (_Float16*)(ws + off); off += (size_t)DIMq * Cq * 2;
  float*    qkvf = (float*)   (ws + off); off += (size_t)Bq * Nq * O3q * 4;
  signed char* q8 = (signed char*)(ws + off); off += (size_t)Bq * Hq * Nq * HDq;
  signed char* k8 = (signed char*)(ws + off); off += (size_t)Bq * Hq * Nq * HDq;
  _Float16* v_t  = (_Float16*)(ws + off); off += (size_t)Bq * Hq * Nq * HDq * 2;
  _Float16* ao_h = (_Float16*)(ws + off); off += (size_t)Bq * Nq * Cq * 2;

  k_cvt_f16<<<4096, 256, 0, stream>>>(x, x_h, Bq * Nq * DIMq);
  k_quant_w<<<2048, 256, 0, stream>>>(qkv_w, qkv_alpha, wq_h, O3q, DIMq);
  k_quant_w<<<1024, 256, 0, stream>>>(proj_w, proj_alpha, wp_h, DIMq, Cq);

  k_gemm_f16<<<dim3(O3q / 128, (Bq * Nq) / 16), dim3(32, 4), 0, stream>>>(
      x_h, wq_h, nullptr, qkvf, Bq * Nq, O3q, DIMq);

  k_qkv_quant<<<(Bq * Hq * Nq) / 256, 256, 0, stream>>>(
      qkvf, bias_q, bias_k, q_alpha, q_zp, k_alpha, k_zp, v_alpha, v_zp, q8, k8, v_t);

  k_attn<<<Bq * Hq * 16, dim3(32, 4), 0, stream>>>(
      q8, k8, v_t, q_alpha, k_alpha, attn_alpha, attn_zp, ao_h);

  k_gemm_f16<<<dim3(DIMq / 128, (Bq * Nq) / 16), dim3(32, 4), 0, stream>>>(
      ao_h, wp_h, proj_b, (float*)d_out, Bq * Nq, DIMq, Cq);
}
